// MinimalKDAAttention_73108933313112
// MI455X (gfx1250) — matrix operand order repaired and checked by reading, not compiled, not run on hardware
//
#include <hip/hip_runtime.h>
#include <hip/hip_bf16.h>
#include <math.h>

typedef __attribute__((ext_vector_type(16))) _Float16 v16h;
typedef __attribute__((ext_vector_type(8)))  float    v8f;
typedef unsigned int u32;

#define BB    2
#define SS    2048
#define HIDC  1024
#define NH    16
#define HD    64
#define KDC   1024
#define MROWS (BB*SS)   /* 4096 */

#define TM 128
#define TN 128
#define TK 32
#define LDP 40          /* LDS row pitch in halfs: 80B = 16B-aligned, conflict-free */

#if defined(__AMDGCN__) && __has_builtin(__builtin_amdgcn_global_load_async_to_lds_b128)
#define KDA_ASYNC 1
#else
#define KDA_ASYNC 0
#endif

#if KDA_ASYNC
// Builtin expects: (global int4*, shared int4*, imm offset, imm cpol)
typedef int kda_v4i __attribute__((vector_size(16)));
typedef __attribute__((address_space(1))) kda_v4i as1_v4i;
typedef __attribute__((address_space(3))) kda_v4i as3_v4i;
#endif

__device__ __forceinline__ void kda_wait_async() {
#if KDA_ASYNC
#if __has_builtin(__builtin_amdgcn_s_wait_asynccnt)
  __builtin_amdgcn_s_wait_asynccnt(0);
#else
  asm volatile("s_wait_asynccnt 0x0" ::: "memory");
#endif
#endif
}

// ---------------- fp32 -> fp16 cast ----------------
__global__ void kda_cast_f16(const float* __restrict__ in, _Float16* __restrict__ out, int n) {
  int i = blockIdx.x * blockDim.x + threadIdx.x;
  if (i < n) out[i] = (_Float16)in[i];
}

// ---------------- fp32 [K][N] -> fp16 transposed [N][K] (LDS-tiled) ----------------
__global__ __launch_bounds__(256) void kda_cast_f16_T(
    const float* __restrict__ in, _Float16* __restrict__ out, int Kd, int Nd) {
  __shared__ float tile[32][33];
  const int nb = blockIdx.x * 32, kb = blockIdx.y * 32;
  const int tx = threadIdx.x & 31, ty = threadIdx.x >> 5;
  #pragma unroll
  for (int i = ty; i < 32; i += 8)
    tile[i][tx] = in[(size_t)(kb + i) * Nd + nb + tx];
  __syncthreads();
  #pragma unroll
  for (int i = ty; i < 32; i += 8)
    out[(size_t)(nb + i) * Kd + kb + tx] = (_Float16)tile[tx][i];
}

// ---------------- WMMA GEMM: C(MxN,f32) = A(MxK,f16,row) * Bt(NxK,f16,row) ----------------
// 256 threads = 8 waves; block tile 128x128; wave = 32 rows x 64 cols (2x4 WMMA accs);
// double-buffered LDS, async global->LDS staging, K-step 32.
__global__ __launch_bounds__(256) void kda_gemm_wmma(
    const _Float16* __restrict__ Ag, const _Float16* __restrict__ Bt,
    float* __restrict__ C, int Mtot, int Ntot, int Ktot) {
  __shared__ _Float16 As[2][TM][LDP];
  __shared__ _Float16 Bs[2][TN][LDP];
  const int tid   = threadIdx.x;
  const int wv    = tid >> 5;
  const int lane  = tid & 31;
  const int wr    = wv & 3;            // row group: rows 32*wr
  const int wc    = wv >> 2;           // col group: cols 64*wc
  const int mBase = blockIdx.y * TM;
  const int nBase = blockIdx.x * TN;

  v8f acc[2][4] = {};

  // Stage one K-tile (A: 128x32, Bt: 128x32) into LDS buffer `buf`.
  auto stage = [&](int buf, int k0) {
    #pragma unroll
    for (int i = 0; i < 2; ++i) {
      const int c   = tid + 256 * i;       // 512 16B-chunks per matrix
      const int row = c >> 2;
      const int ch  = (c & 3) * 8;         // half offset within row
      const _Float16* ga = Ag + (size_t)(mBase + row) * Ktot + k0 + ch;
      const _Float16* gb = Bt + (size_t)(nBase + row) * Ktot + k0 + ch;
      if (k0 + 2 * TK < Ktot) {            // prefetch tile after next
        __builtin_prefetch(ga + 2 * TK, 0, 0);
        __builtin_prefetch(gb + 2 * TK, 0, 0);
      }
#if KDA_ASYNC
      __builtin_amdgcn_global_load_async_to_lds_b128(
          (as1_v4i*)ga, (as3_v4i*)&As[buf][row][ch], 0, 0);
      __builtin_amdgcn_global_load_async_to_lds_b128(
          (as1_v4i*)gb, (as3_v4i*)&Bs[buf][row][ch], 0, 0);
#else
      uint4 va = *(const uint4*)ga;
      uint4 vb = *(const uint4*)gb;
      *(uint4*)&As[buf][row][ch] = va;
      *(uint4*)&Bs[buf][row][ch] = vb;
#endif
    }
  };

  // One K-step of WMMA work on buffer `buf`.
  auto compute = [&](int buf) {
    union UF { v16h h; u32 u[8]; };
    const int rsel = lane & 15;
    const int pbA  = (lane >> 4) << 3;     // A pair offset
    const int pbB  = (lane >> 4) << 3;     // B pair offset
    UF af[2];
    #pragma unroll
    for (int t = 0; t < 2; ++t) {
      const u32* arow = (const u32*)&As[buf][wr * 32 + t * 16 + rsel][0];
      #pragma unroll
      for (int r = 0; r < 4; ++r) {
        af[t].u[r]     = arow[pbA + r];
        af[t].u[4 + r] = arow[4 + pbA + r];
      }
    }
    #pragma unroll
    for (int cg = 0; cg < 4; ++cg) {
      UF bf;
      const u32* brow = (const u32*)&Bs[buf][wc * 64 + cg * 16 + rsel][0];
      #pragma unroll
      for (int r = 0; r < 8; ++r) bf.u[r] = brow[pbB + r];
      #pragma unroll
      for (int t = 0; t < 2; ++t)
        acc[t][cg] = __builtin_amdgcn_wmma_f32_16x16x32_f16(
            false, af[t].h, false, bf.h, (short)0, acc[t][cg], false, false);
    }
  };

  stage(0, 0);
  kda_wait_async();
  __syncthreads();
  int buf = 0;
  for (int k0 = 0; k0 < Ktot; k0 += TK) {
    if (k0 + TK < Ktot) stage(buf ^ 1, k0 + TK);
    compute(buf);
    kda_wait_async();
    __syncthreads();
    buf ^= 1;
  }

  // C/D layout: lanes 0-15 -> M=r, lanes 16-31 -> M=r+8; N = lane&15
  const int lrow = (lane >> 4) << 3;
  const int lcol = lane & 15;
  #pragma unroll
  for (int t = 0; t < 2; ++t)
    #pragma unroll
    for (int cg = 0; cg < 4; ++cg)
      #pragma unroll
      for (int r = 0; r < 8; ++r) {
        int row = mBase + wr * 32 + t * 16 + r + lrow;
        int col = nBase + wc * 64 + cg * 16 + lcol;
        C[(size_t)row * Ntot + col] = acc[t][cg][r];
      }
}

// ---------------- beta = sigmoid(x @ Wb), N=16 (tiny) ----------------
__global__ void kda_beta(const float* __restrict__ x, const float* __restrict__ Wb,
                         float* __restrict__ beta) {
  int i = blockIdx.x * blockDim.x + threadIdx.x;
  if (i >= MROWS * NH) return;
  int row = i / NH, h = i % NH;
  const float* xr = x + (size_t)row * HIDC;
  float acc = 0.f;
  for (int kk = 0; kk < HIDC; ++kk) acc = fmaf(xr[kk], Wb[kk * NH + h], acc);
  beta[i] = 1.f / (1.f + expf(-acc));
}

// ---------------- silu then L2-normalize each 64-wide row (one wave32 per row) ----------------
__global__ void kda_silu_l2norm(float* __restrict__ t, int nrows) {
  int wid  = blockIdx.x * (blockDim.x >> 5) + (threadIdx.x >> 5);
  int lane = threadIdx.x & 31;
  if (wid >= nrows) return;
  float* row = t + (size_t)wid * HD;
  float a = row[lane], b = row[lane + 32];
  a = a / (1.f + expf(-a));
  b = b / (1.f + expf(-b));
  float ss = a * a + b * b;
  #pragma unroll
  for (int off = 16; off > 0; off >>= 1) ss += __shfl_xor(ss, off, 32);
  float inv = 1.f / fmaxf(sqrtf(ss), 1e-12f);
  row[lane]      = a * inv;
  row[lane + 32] = b * inv;
}

// ---------------- elementwise silu ----------------
__global__ void kda_silu(float* __restrict__ t, int n) {
  int i = blockIdx.x * blockDim.x + threadIdx.x;
  if (i < n) { float v = t[i]; t[i] = v / (1.f + expf(-v)); }
}

// ---------------- t = sigmoid(t + sgn * bias[col % KD]) ----------------
__global__ void kda_sigmoid_bias(float* __restrict__ t, const float* __restrict__ bias,
                                 float sgn, int n) {
  int i = blockIdx.x * blockDim.x + threadIdx.x;
  if (i < n) {
    float v = t[i] + sgn * bias[i % KDC];
    t[i] = 1.f / (1.f + expf(-v));
  }
}

// ---------------- sequential gated scan ----------------
// One block (64 thr, 2 waves) per (b,h); thread e owns state column S[:,e] in VGPRs.
// q/k broadcast via wave32 shuffles: no LDS, no barriers.
__global__ __launch_bounds__(64) void kda_scan(
    const float* __restrict__ q, const float* __restrict__ k,
    const float* __restrict__ v, const float* __restrict__ g,
    const float* __restrict__ beta, const float* __restrict__ A_log,
    float* __restrict__ o) {
  const int bh = blockIdx.x;
  const int b = bh / NH, h = bh % NH;
  const int e = threadIdx.x;
  const int lane = e & 31;
  const float Ah = expf(-expf(A_log[h]));
  float st[HD];
  #pragma unroll
  for (int d = 0; d < HD; ++d) st[d] = 0.f;
  for (int s = 0; s < SS; ++s) {
    const size_t base = (size_t)(b * SS + s) * KDC + h * HD;
    const float ql = q[base + lane],      kl = k[base + lane];
    const float qh = q[base + 32 + lane], kh = k[base + 32 + lane];
    const float ve = v[base + e];
    const float ge = g[base + e];
    const float bt = beta[(size_t)(b * SS + s) * NH + h];
    const float bv = bt * ve;
    float accv = 0.f;
    #pragma unroll
    for (int d = 0; d < 32; ++d) {
      const float qd = __shfl(ql, d, 32);
      const float kd = __shfl(kl, d, 32);
      accv  = fmaf(qd, st[d], accv);
      st[d] = fmaf(Ah, st[d], kd * bv);
    }
    #pragma unroll
    for (int d = 0; d < 32; ++d) {
      const float qd = __shfl(qh, d, 32);
      const float kd = __shfl(kh, d, 32);
      accv       = fmaf(qd, st[32 + d], accv);
      st[32 + d] = fmaf(Ah, st[32 + d], kd * bv);
    }
    o[base + e] = accv * ge;
  }
}

// ---------------- RMSNorm over 64 + norm_w + gate multiply (one wave32 per row) ----------------
__global__ void kda_rmsnorm_gate(float* __restrict__ o, const float* __restrict__ gate,
                                 const float* __restrict__ norm_w, int nrows) {
  int wid  = blockIdx.x * (blockDim.x >> 5) + (threadIdx.x >> 5);
  int lane = threadIdx.x & 31;
  if (wid >= nrows) return;
  float* row = o + (size_t)wid * HD;
  const float* gr = gate + (size_t)wid * HD;
  float a = row[lane], b = row[lane + 32];
  float ss = a * a + b * b;
  #pragma unroll
  for (int off = 16; off > 0; off >>= 1) ss += __shfl_xor(ss, off, 32);
  float inv = rsqrtf(ss * (1.f / HD) + 1e-5f);
  row[lane]      = a * inv * norm_w[lane]      * gr[lane];
  row[lane + 32] = b * inv * norm_w[lane + 32] * gr[lane + 32];
}

extern "C" void kernel_launch(void* const* d_in, const int* in_sizes, int n_in,
                              void* d_out, int out_size, void* d_ws, size_t ws_size,
                              hipStream_t stream) {
  (void)in_sizes; (void)n_in; (void)out_size; (void)ws_size;
  const float* x      = (const float*)d_in[0];
  const float* Wq     = (const float*)d_in[1];
  const float* Wk     = (const float*)d_in[2];
  const float* Wv     = (const float*)d_in[3];
  const float* Wf     = (const float*)d_in[4];
  const float* Wb     = (const float*)d_in[5];
  const float* A_log  = (const float*)d_in[6];
  const float* dt_b   = (const float*)d_in[7];
  const float* Wg     = (const float*)d_in[8];
  const float* bg     = (const float*)d_in[9];
  const float* Wo     = (const float*)d_in[10];
  const float* norm_w = (const float*)d_in[11];

  char* ws = (char*)d_ws;
  size_t off = 0;
  auto take = [&](size_t bytes) -> char* {
    char* p = ws + off;
    off = (off + bytes + 255) & ~(size_t)255;
    return p;
  };

  _Float16* xh   = (_Float16*)take((size_t)MROWS * HIDC * 2);
  _Float16* WqT  = (_Float16*)take((size_t)HIDC * KDC * 2);
  _Float16* WkT  = (_Float16*)take((size_t)HIDC * KDC * 2);
  _Float16* WvT  = (_Float16*)take((size_t)HIDC * KDC * 2);
  _Float16* WfT  = (_Float16*)take((size_t)HIDC * KDC * 2);
  _Float16* WgT  = (_Float16*)take((size_t)HIDC * KDC * 2);
  _Float16* WoT  = (_Float16*)take((size_t)KDC * HIDC * 2);
  float* qb    = (float*)take((size_t)MROWS * KDC * 4);
  float* kb    = (float*)take((size_t)MROWS * KDC * 4);
  float* vb    = (float*)take((size_t)MROWS * KDC * 4);
  float* gb    = (float*)take((size_t)MROWS * KDC * 4);
  float* gateb = (float*)take((size_t)MROWS * KDC * 4);
  float* ob    = (float*)take((size_t)MROWS * KDC * 4);
  float* betab = (float*)take((size_t)MROWS * NH * 4);
  _Float16* oh = (_Float16*)take((size_t)MROWS * KDC * 2);

  const int nxe = MROWS * HIDC;
  kda_cast_f16<<<(nxe + 255) / 256, 256, 0, stream>>>(x, xh, nxe);
  dim3 tg(KDC / 32, HIDC / 32);   // weights are [HID][KD] -> transposed [KD][HID]
  kda_cast_f16_T<<<tg, 256, 0, stream>>>(Wq, WqT, HIDC, KDC);
  kda_cast_f16_T<<<tg, 256, 0, stream>>>(Wk, WkT, HIDC, KDC);
  kda_cast_f16_T<<<tg, 256, 0, stream>>>(Wv, WvT, HIDC, KDC);
  kda_cast_f16_T<<<tg, 256, 0, stream>>>(Wf, WfT, HIDC, KDC);
  kda_cast_f16_T<<<tg, 256, 0, stream>>>(Wg, WgT, HIDC, KDC);
  dim3 tgo(HIDC / 32, KDC / 32);  // Wo is [KD][HID] -> transposed [HID][KD]
  kda_cast_f16_T<<<tgo, 256, 0, stream>>>(Wo, WoT, KDC, HIDC);

  dim3 gg(KDC / TN, MROWS / TM);  // (8, 32)
  kda_gemm_wmma<<<gg, 256, 0, stream>>>(xh, WqT, qb,    MROWS, KDC, HIDC);
  kda_gemm_wmma<<<gg, 256, 0, stream>>>(xh, WkT, kb,    MROWS, KDC, HIDC);
  kda_gemm_wmma<<<gg, 256, 0, stream>>>(xh, WvT, vb,    MROWS, KDC, HIDC);
  kda_gemm_wmma<<<gg, 256, 0, stream>>>(xh, WfT, gb,    MROWS, KDC, HIDC);
  kda_gemm_wmma<<<gg, 256, 0, stream>>>(xh, WgT, gateb, MROWS, KDC, HIDC);

  kda_beta<<<(MROWS * NH + 255) / 256, 256, 0, stream>>>(x, Wb, betab);

  const int nrows = MROWS * NH;        // 65536 rows of 64
  const int rblk  = (nrows + 3) / 4;   // 4 wave-rows per 128-thread block
  kda_silu_l2norm<<<rblk, 128, 0, stream>>>(qb, nrows);
  kda_silu_l2norm<<<rblk, 128, 0, stream>>>(kb, nrows);

  const int nel = MROWS * KDC;
  kda_silu<<<(nel + 255) / 256, 256, 0, stream>>>(vb, nel);
  kda_sigmoid_bias<<<(nel + 255) / 256, 256, 0, stream>>>(gb,    dt_b, -1.f, nel);
  kda_sigmoid_bias<<<(nel + 255) / 256, 256, 0, stream>>>(gateb, bg,   +1.f, nel);

  kda_scan<<<BB * NH, 64, 0, stream>>>(qb, kb, vb, gb, betab, A_log, ob);

  kda_rmsnorm_gate<<<rblk, 128, 0, stream>>>(ob, gateb, norm_w, nrows);

  kda_cast_f16<<<(nel + 255) / 256, 256, 0, stream>>>(ob, oh, nel);
  dim3 go(HIDC / TN, MROWS / TM);
  kda_gemm_wmma<<<go, 256, 0, stream>>>(oh, WoT, (float*)d_out, MROWS, HIDC, KDC);
}
